// Encoder_Layer_59614146068573
// MI455X (gfx1250) — matrix operand order repaired and checked by reading, not compiled, not run on hardware
//
#include <hip/hip_runtime.h>

// ---------------- problem constants (from reference) ----------------
constexpr int BB  = 4;      // batch
constexpr int SS  = 2048;   // sequence
constexpr int EE  = 512;    // embedding
constexpr int HH  = 8;      // heads
constexpr int DK  = 64;     // head dim (K and V)
constexpr int MID = 1024;   // FFN hidden
constexpr int BSR = BB * SS;          // 8192 token rows
constexpr float LNEPS = 1e-5f;

// GEMM tiling: 128x64 block tile, K-step 32, 8 waves (4x2), 4 WMMA/wave/tile
constexpr int TM = 128, TN = 64, TK = 32;

typedef __attribute__((ext_vector_type(16))) __bf16 v16bf;
typedef __attribute__((ext_vector_type(8)))  float  v8f;

union FragBF { v16bf v; uint4 u[2]; };

__device__ __forceinline__ unsigned short f2bf(float f) {
  unsigned int u = __float_as_uint(f);
  u += 0x7FFFu + ((u >> 16) & 1u);          // round-to-nearest-even
  return (unsigned short)(u >> 16);
}

// ---- CDNA5 async global->LDS (ASYNCcnt) --------------------------------
// Generic pointers to LDS carry the LDS offset in their low 32 bits.
__device__ __forceinline__ void async_b128(void* lds, const void* gaddr) {
  unsigned loff = (unsigned)(size_t)lds;
  asm volatile("global_load_async_to_lds_b128 %0, %1, off"
               :: "v"(loff), "v"(gaddr) : "memory");
}
__device__ __forceinline__ void wait_async0() { asm volatile("s_wait_asynccnt 0x0" ::: "memory"); }
__device__ __forceinline__ void wait_async1() { asm volatile("s_wait_asynccnt 0x1" ::: "memory"); }
__device__ __forceinline__ void wait_async3() { asm volatile("s_wait_asynccnt 0x3" ::: "memory"); }

// epilogue modes
enum { EPI_QKV = 0, EPI_SCORES = 1, EPI_AV = 2, EPI_BIASRES = 3, EPI_RELU = 4, EPI_VT = 5 };

// C[M,N](f32 acc) = A[M,K] x Bt, with Bt given [N][K] row-major (weights are
// pre-transposed on device; K^T and V^T come naturally in that layout).
// A is bf16 (async-staged) or f32 (converted to bf16 while staging, AV only).
template<int EPI, bool AF32>
__global__ __launch_bounds__(256)
void gemm_wmma_bf16(const void* __restrict__ Ap,
                    const unsigned short* __restrict__ Bp,
                    const float* __restrict__ bias,
                    const float* __restrict__ resid,
                    const int* __restrict__ mask,
                    float* __restrict__ outF,
                    unsigned short* __restrict__ outB,
                    int M, int N, int K, int lda, int ldb,
                    long aBatch, long bBatch)
{
  __shared__ unsigned short As[2][TM][TK];   // 16 KB (double buffered)
  __shared__ unsigned short Bs[2][TN][TK];   //  8 KB, stored [n][k]

  const int tid  = threadIdx.x;
  const int lane = tid & 31;
  const int wave = tid >> 5;
  const int wy   = wave >> 1;               // 0..3 : 32-row band
  const int wx   = wave & 1;                // 0..1 : 32-col band
  const int z    = blockIdx.z;
  const int m0   = blockIdx.y * TM;
  const int n0   = blockIdx.x * TN;

  const float*          Af = (const float*)Ap          + (AF32 ? (long)z * aBatch : 0);
  const unsigned short* Ab = (const unsigned short*)Ap + (AF32 ? 0 : (long)z * aBatch);
  const unsigned short* Bb = Bp + (long)z * bBatch;

  auto stage = [&](int k0, int buf) {
    { // B tile 64x32: one async b128 per thread
      const int r = tid >> 2, c = (tid & 3) * 8;
      async_b128(&Bs[buf][r][c], Bb + (long)(n0 + r) * ldb + (k0 + c));
    }
    // A tile 128x32: two 16B chunks per thread
    if constexpr (AF32) {
      #pragma unroll
      for (int q = 0; q < 2; ++q) {
        const int ch = tid * 2 + q;
        const int r = ch >> 2, c = (ch & 3) * 8;
        const float* s = Af + (long)(m0 + r) * lda + (k0 + c);
        float4 f0 = ((const float4*)s)[0];
        float4 f1 = ((const float4*)s)[1];
        unsigned short t8[8] = { f2bf(f0.x), f2bf(f0.y), f2bf(f0.z), f2bf(f0.w),
                                 f2bf(f1.x), f2bf(f1.y), f2bf(f1.z), f2bf(f1.w) };
        *(uint4*)&As[buf][r][c] = *(const uint4*)t8;
      }
    } else {
      #pragma unroll
      for (int q = 0; q < 2; ++q) {
        const int ch = tid * 2 + q;
        const int r = ch >> 2, c = (ch & 3) * 8;
        async_b128(&As[buf][r][c], Ab + (long)(m0 + r) * lda + (k0 + c));
      }
    }
  };

  v8f acc[2][2] = {};
  const int T = K / TK;
  stage(0, 0);
  for (int i = 0; i < T; ++i) {
    const int cur = i & 1;
    if (i + 1 < T) {
      stage((i + 1) * TK, cur ^ 1);            // prefetch next tile (async)
      if constexpr (AF32) wait_async1(); else wait_async3();
    } else {
      wait_async0();
    }
    __syncthreads();                           // tile i visible to all waves

    const int khalf = lane >> 4;
    const int l15   = lane & 15;
    FragBF a[2], b[2];
    #pragma unroll
    for (int ii = 0; ii < 2; ++ii) {           // A 16x32 layout: K groups of 8/half
      const int mr = wy * 32 + ii * 16 + l15;
      a[ii].u[0] = *(const uint4*)&As[cur][mr][khalf * 8];
      a[ii].u[1] = *(const uint4*)&As[cur][mr][16 + khalf * 8];
    }
    #pragma unroll
    for (int jj = 0; jj < 2; ++jj) {           // B: K groups of 8/half (same as A)
      const int nr = wx * 32 + jj * 16 + l15;
      b[jj].u[0] = *(const uint4*)&Bs[cur][nr][khalf * 8];
      b[jj].u[1] = *(const uint4*)&Bs[cur][nr][16 + khalf * 8];
    }
    #pragma unroll
    for (int ii = 0; ii < 2; ++ii)
      #pragma unroll
      for (int jj = 0; jj < 2; ++jj)
        acc[ii][jj] = __builtin_amdgcn_wmma_f32_16x16x32_bf16(
            false, a[ii].v, false, b[jj].v, (short)0, acc[ii][jj], false, false);
    __syncthreads();                           // protect buffer being refilled
  }

  // epilogue: C VGPR r -> lanes 0-15 row r, lanes 16-31 row r+8
  const int lm = (lane >> 4) * 8;
  const int ln = lane & 15;
  #pragma unroll
  for (int ii = 0; ii < 2; ++ii)
  #pragma unroll
  for (int jj = 0; jj < 2; ++jj)
  #pragma unroll
  for (int r = 0; r < 8; ++r) {
    const int m = m0 + wy * 32 + ii * 16 + lm + r;
    const int n = n0 + wx * 32 + jj * 16 + ln;
    float val = acc[ii][jj][r];
    if constexpr (EPI == EPI_QKV) {
      val += bias[n];
      const int bb = m >> 11, s = m & (SS - 1);   // m = b*S + s
      const int h  = n >> 6,  d = n & (DK - 1);   // n = h*DK + d
      outB[(((long)(bb * HH + h) * SS + s) << 6) + d] = f2bf(val);
    } else if constexpr (EPI == EPI_VT) {         // V stored [B,H,D,S]
      val += bias[n];
      const int bb = m >> 11, s = m & (SS - 1);
      const int h  = n >> 6,  d = n & (DK - 1);
      outB[((long)(bb * HH + h) * DK + d) * SS + s] = f2bf(val);
    } else if constexpr (EPI == EPI_SCORES) {
      const int bb = z >> 3;                      // z = b*H + h
      val *= 0.125f;                              // 1/sqrt(DK)
      if (mask[((long)bb * SS + m) * SS + n] == 0) val = -1e9f;
      outF[((long)z * SS + m) * SS + n] = val;
    } else if constexpr (EPI == EPI_AV) {
      const int bb = z >> 3, h = z & (HH - 1);
      outB[(long)(bb * SS + m) * (HH * DK) + h * DK + n] = f2bf(val);
    } else if constexpr (EPI == EPI_BIASRES) {
      val += bias[n] + resid[(long)m * N + n];
      outF[(long)m * N + n] = val;
    } else {  // EPI_RELU
      val += bias[n];
      outB[(long)m * N + n] = f2bf(fmaxf(val, 0.f));
    }
  }
}

// row softmax over S=2048 cols, in place (these rows ARE the attn_w output)
__global__ __launch_bounds__(256)
void softmax_rows(float* __restrict__ attn)
{
  __shared__ float red[256];
  const int t = threadIdx.x;
  float* p = attn + (long)blockIdx.x * SS + t * 8;
  float4 v0 = ((const float4*)p)[0];
  float4 v1 = ((const float4*)p)[1];
  float mx = fmaxf(fmaxf(fmaxf(v0.x, v0.y), fmaxf(v0.z, v0.w)),
                   fmaxf(fmaxf(v1.x, v1.y), fmaxf(v1.z, v1.w)));
  red[t] = mx; __syncthreads();
  for (int o = 128; o > 0; o >>= 1) { if (t < o) red[t] = fmaxf(red[t], red[t + o]); __syncthreads(); }
  const float rmax = red[0]; __syncthreads();
  float e[8];
  e[0]=__expf(v0.x-rmax); e[1]=__expf(v0.y-rmax); e[2]=__expf(v0.z-rmax); e[3]=__expf(v0.w-rmax);
  e[4]=__expf(v1.x-rmax); e[5]=__expf(v1.y-rmax); e[6]=__expf(v1.z-rmax); e[7]=__expf(v1.w-rmax);
  red[t] = ((e[0]+e[1])+(e[2]+e[3])) + ((e[4]+e[5])+(e[6]+e[7])); __syncthreads();
  for (int o = 128; o > 0; o >>= 1) { if (t < o) red[t] += red[t + o]; __syncthreads(); }
  const float inv = 1.0f / red[0];
  ((float4*)p)[0] = make_float4(e[0]*inv, e[1]*inv, e[2]*inv, e[3]*inv);
  ((float4*)p)[1] = make_float4(e[4]*inv, e[5]*inv, e[6]*inv, e[7]*inv);
}

// row LayerNorm over E=512 cols; writes f32 and optional bf16 copy
__global__ __launch_bounds__(256)
void layernorm_rows(const float* __restrict__ in, const float* __restrict__ gamma,
                    const float* __restrict__ beta, float* __restrict__ outF,
                    unsigned short* __restrict__ outB)
{
  __shared__ float red[256];
  const int t = threadIdx.x;
  const long row = blockIdx.x;
  const float2 v = ((const float2*)(in + row * EE))[t];
  red[t] = v.x + v.y; __syncthreads();
  for (int o = 128; o > 0; o >>= 1) { if (t < o) red[t] += red[t + o]; __syncthreads(); }
  const float mean = red[0] * (1.0f / EE); __syncthreads();
  const float dx = v.x - mean, dy = v.y - mean;
  red[t] = dx * dx + dy * dy; __syncthreads();
  for (int o = 128; o > 0; o >>= 1) { if (t < o) red[t] += red[t + o]; __syncthreads(); }
  const float inv = rsqrtf(red[0] * (1.0f / EE) + LNEPS);
  const int n = t * 2;
  const float o0 = dx * inv * gamma[n]     + beta[n];
  const float o1 = dy * inv * gamma[n + 1] + beta[n + 1];
  ((float2*)(outF + row * EE))[t] = make_float2(o0, o1);
  if (outB) { outB[row * EE + n] = f2bf(o0); outB[row * EE + n + 1] = f2bf(o1); }
}

__global__ __launch_bounds__(256)
void cvt_f32_to_bf16(const float* __restrict__ in, unsigned short* __restrict__ out, int n)
{
  const int i = blockIdx.x * 256 + threadIdx.x;
  if (i < n) out[i] = f2bf(in[i]);
}

// f32 [K][N] -> bf16 [N][K] (so every GEMM B operand is [N][K] row-major)
__global__ __launch_bounds__(256)
void cvt_transpose_bf16(const float* __restrict__ in, unsigned short* __restrict__ out,
                        int Kd, int Nd)
{
  __shared__ float t[32][33];
  const int k0 = blockIdx.y * 32, n0 = blockIdx.x * 32;
  const int tx = threadIdx.x & 31, ty = threadIdx.x >> 5;
  #pragma unroll
  for (int j = ty; j < 32; j += 8) t[j][tx] = in[(long)(k0 + j) * Nd + (n0 + tx)];
  __syncthreads();
  #pragma unroll
  for (int j = ty; j < 32; j += 8) out[(long)(n0 + j) * Kd + (k0 + tx)] = f2bf(t[tx][j]);
}

extern "C" void kernel_launch(void* const* d_in, const int* in_sizes, int n_in,
                              void* d_out, int out_size, void* d_ws, size_t ws_size,
                              hipStream_t stream)
{
  const float* x    = (const float*)d_in[0];
  const int*   mask = (const int*)  d_in[1];
  const float* Wq = (const float*)d_in[2];  const float* bq  = (const float*)d_in[3];
  const float* Wk = (const float*)d_in[4];  const float* bk  = (const float*)d_in[5];
  const float* Wv = (const float*)d_in[6];  const float* bv  = (const float*)d_in[7];
  const float* Wo = (const float*)d_in[8];  const float* bo  = (const float*)d_in[9];
  const float* g1 = (const float*)d_in[10]; const float* be1 = (const float*)d_in[11];
  const float* W1 = (const float*)d_in[12]; const float* b1  = (const float*)d_in[13];
  const float* W2 = (const float*)d_in[14]; const float* b2  = (const float*)d_in[15];
  const float* g2 = (const float*)d_in[16]; const float* be2 = (const float*)d_in[17];

  float* outp = (float*)d_out;               // [B,S,E]
  float* attn = outp + (long)BSR * EE;       // [B,H,S,S] (2nd tuple output)

  size_t off = 0;
  auto take = [&](size_t bytes) -> void* {
    void* p = (char*)d_ws + off;
    off += (bytes + 255) & ~(size_t)255;
    return p;
  };
  unsigned short* xb  = (unsigned short*)take((size_t)BSR * EE  * 2);
  unsigned short* wqt = (unsigned short*)take((size_t)EE  * EE  * 2);  // [N][K]
  unsigned short* wkt = (unsigned short*)take((size_t)EE  * EE  * 2);
  unsigned short* wvt = (unsigned short*)take((size_t)EE  * EE  * 2);
  unsigned short* wot = (unsigned short*)take((size_t)EE  * EE  * 2);
  unsigned short* w1t = (unsigned short*)take((size_t)EE  * MID * 2);
  unsigned short* w2t = (unsigned short*)take((size_t)MID * EE  * 2);
  unsigned short* qb  = (unsigned short*)take((size_t)BSR * EE  * 2);  // [B,H,S,DK]
  unsigned short* kb  = (unsigned short*)take((size_t)BSR * EE  * 2);  // [B,H,S,DK]
  unsigned short* vb  = (unsigned short*)take((size_t)BSR * EE  * 2);  // [B,H,DK,S]
  unsigned short* ab  = (unsigned short*)take((size_t)BSR * EE  * 2);  // [B,S,H*DV]
  unsigned short* hb  = (unsigned short*)take((size_t)BSR * EE  * 2);
  unsigned short* fb  = (unsigned short*)take((size_t)BSR * MID * 2);
  float* tmp = (float*)take((size_t)BSR * EE * 4);
  float* hf  = (float*)take((size_t)BSR * EE * 4);

  cvt_f32_to_bf16<<<(BSR * EE + 255) / 256, 256, 0, stream>>>(x, xb, BSR * EE);
  auto cvtT = [&](const float* s, unsigned short* d, int Kd, int Nd) {
    cvt_transpose_bf16<<<dim3(Nd / 32, Kd / 32), 256, 0, stream>>>(s, d, Kd, Nd);
  };
  cvtT(Wq, wqt, EE, EE);  cvtT(Wk, wkt, EE, EE);  cvtT(Wv, wvt, EE, EE);
  cvtT(Wo, wot, EE, EE);  cvtT(W1, w1t, EE, MID); cvtT(W2, w2t, MID, EE);

  // QKV projections: [BSR,E] x Wt (+bias) -> bf16, V in transposed layout
  {
    dim3 g(EE / TN, BSR / TM, 1);
    gemm_wmma_bf16<EPI_QKV, false><<<g, 256, 0, stream>>>(
        xb, wqt, bq, nullptr, nullptr, nullptr, qb, BSR, EE, EE, EE, EE, 0, 0);
    gemm_wmma_bf16<EPI_QKV, false><<<g, 256, 0, stream>>>(
        xb, wkt, bk, nullptr, nullptr, nullptr, kb, BSR, EE, EE, EE, EE, 0, 0);
    gemm_wmma_bf16<EPI_VT, false><<<g, 256, 0, stream>>>(
        xb, wvt, bv, nullptr, nullptr, nullptr, vb, BSR, EE, EE, EE, EE, 0, 0);
  }
  // scores per (b,h): q [S,DK] x k^T (k rows are already [N][K])
  {
    dim3 g(SS / TN, SS / TM, BB * HH);
    gemm_wmma_bf16<EPI_SCORES, false><<<g, 256, 0, stream>>>(
        qb, kb, nullptr, nullptr, mask, attn, nullptr,
        SS, SS, DK, DK, DK, (long)SS * DK, (long)SS * DK);
  }
  softmax_rows<<<BB * HH * SS, 256, 0, stream>>>(attn);
  // attn(f32 -> bf16 while staging) @ v  (v stored [D][S] => [N][K])
  {
    dim3 g(DK / TN, SS / TM, BB * HH);
    gemm_wmma_bf16<EPI_AV, true><<<g, 256, 0, stream>>>(
        attn, vb, nullptr, nullptr, nullptr, nullptr, ab,
        SS, DK, SS, SS, SS, (long)SS * SS, (long)DK * SS);
  }
  // out-proj + bias + residual(x) -> tmp f32 ; LN1 -> hf/hb
  {
    dim3 g(EE / TN, BSR / TM, 1);
    gemm_wmma_bf16<EPI_BIASRES, false><<<g, 256, 0, stream>>>(
        ab, wot, bo, x, nullptr, tmp, nullptr, BSR, EE, EE, EE, EE, 0, 0);
  }
  layernorm_rows<<<BSR, 256, 0, stream>>>(tmp, g1, be1, hf, hb);
  // FFN1 + ReLU -> fb bf16
  {
    dim3 g(MID / TN, BSR / TM, 1);
    gemm_wmma_bf16<EPI_RELU, false><<<g, 256, 0, stream>>>(
        hb, w1t, b1, nullptr, nullptr, nullptr, fb, BSR, MID, EE, EE, EE, 0, 0);
  }
  // FFN2 + bias + residual(h) -> tmp ; LN2 -> final out
  {
    dim3 g(EE / TN, BSR / TM, 1);
    gemm_wmma_bf16<EPI_BIASRES, false><<<g, 256, 0, stream>>>(
        fb, w2t, b2, hf, nullptr, tmp, nullptr, BSR, EE, MID, MID, MID, 0, 0);
  }
  layernorm_rows<<<BSR, 256, 0, stream>>>(tmp, g2, be2, outp, nullptr);

  (void)in_sizes; (void)n_in; (void)out_size; (void)ws_size;
}
